// RnActor_22531398434859
// MI455X (gfx1250) — matrix operand order repaired and checked by reading, not compiled, not run on hardware
//
#include <hip/hip_runtime.h>
#include <hip/hip_bf16.h>

typedef __attribute__((ext_vector_type(16))) _Float16 v16h;
typedef __attribute__((ext_vector_type(8)))  _Float16 v8h;
typedef __attribute__((ext_vector_type(8)))  float    v8f;

#define NB_OBJ 5
#define DIM_OBJ 15
#define NPERM 20
#define TB 8                 // batch rows per workgroup
#define MROWS (TB*NPERM)     // 160 edge rows
#define BATCH 32768

// ---- d_ws fragment offsets (in halves) ----
#define WF_W1 0          // 16 tiles  (K=32pad from 19, N=256)
#define WF_W2 8192       // 64 tiles  (K=256, N=128)
#define WF_WQ 40960      // 32 tiles  (K=128, N=128)
#define WF_WK 57344
#define WF_WV 73728
#define WF_WR 90112      // 64 tiles  (K=128, N=256)   -> total 122880 halves = 245760 B of d_ws

// ---- LDS layout (bytes), total 253952 (248 KB) ----
#define H_OFF  0          // h: 160x256 f16 (81920) ; later: Q 8x32x128 f16 @0 (65536)
#define SC_OFF 65536      // scores: 8x20x20 f32 (12800)  (inside dead upper h region)
#define X_OFF  81920      // inp 160x32 f16 then x 160x128 f16 (40960); later attn/pooled/hr
#define AT_OFF 81920      // attn A-tiles: 8x32x32 f16 (16384)
#define PA_OFF 98304      // pooled f16: 16x128 (4096)
#define HR_OFF 102400     // hr f16: 16x256 (8192)
#define K_OFF  122880     // K: 8x32x128 f16 (65536)
#define VT_OFF 188416     // V^T: 8x128x32 f16 (65536)
#define LDS_BYTES 253952

__device__ __forceinline__ v8f wmma16(v16h a, v16h b, v8f c) {
  return __builtin_amdgcn_wmma_f32_16x16x32_f16(false, a, false, b, (short)0, c, false, false);
}

// A fragment (16x32 f16) from row-major LDS. row = ptr to this lane's M row.
// lanes 0-15 hold K {0..15}+koff, lanes 16-31 hold K {16..31}+koff.
__device__ __forceinline__ v16h load_a(const _Float16* row, int koff, int lane) {
  int k0 = koff + ((lane >> 4) & 1) * 16;
  v8h lo = *(const v8h*)(row + k0);        // ds_load_b128
  v8h hi = *(const v8h*)(row + k0 + 8);    // ds_load_b128
  v16h r;
#pragma unroll
  for (int i = 0; i < 8; ++i) { r[i] = lo[i]; r[i + 8] = hi[i]; }
  return r;
}

// B fragment (32x16 f16) from row-major LDS where rows are the N columns of the
// mathematical B (i.e. buffer[n][k]). lane n=l&15, K-half chosen by l>=16.
__device__ __forceinline__ v16h load_b_rows(const _Float16* nrow, int koff, int lane) {
  const v16h* p = (const v16h*)(nrow + koff + ((lane >> 4) & 1) * 16);
  return *p;                                // 2x ds_load_b128
}

// ---------------- weight packing: f32 row-major [K][N] -> per-tile lane images
__global__ void pack_weights(const float* __restrict__ W, _Float16* __restrict__ dst,
                             int Ksrc, int N, int Ntiles) {
  int tile = blockIdx.x;
  int kt = tile / Ntiles, nt = tile % Ntiles;
  int lane = threadIdx.x;
  int n = nt * 16 + (lane & 15);
  int kb = kt * 32 + ((lane >> 4) & 1) * 16;
  _Float16* o = dst + (size_t)tile * 512 + lane * 16;
#pragma unroll
  for (int h = 0; h < 16; ++h) {
    int k = kb + h;
    float v = (k < Ksrc) ? W[(size_t)k * N + n] : 0.0f;
    o[h] = (_Float16)v;
  }
}

// ---------------- fused actor kernel ----------------
__global__ __launch_bounds__(256, 1)
void actor_fused(const float* __restrict__ obs, const float* __restrict__ ag,
                 const float* __restrict__ g,
                 const float* __restrict__ b1, const float* __restrict__ b2,
                 const float* __restrict__ bq, const float* __restrict__ bk,
                 const float* __restrict__ bv, const float* __restrict__ br,
                 const float* __restrict__ Wm, const float* __restrict__ bm,
                 const float* __restrict__ Ws, const float* __restrict__ bs,
                 const int* __restrict__ edges, const int* __restrict__ pred_ids,
                 const _Float16* __restrict__ wf, float* __restrict__ out) {
  __shared__ __align__(16) unsigned char smem[LDS_BYTES];
  const int tid = threadIdx.x;
  const int w = tid >> 5, lane = tid & 31;

  // ---- stage 0: build 160x32 (f16, zero-padded K) edge inputs ----
  if (tid < MROWS) {
    int bi = tid / NPERM, p = tid % NPERM;
    int b = blockIdx.x * TB + bi;
    const float* ob = obs + (size_t)b * (10 + NB_OBJ * DIM_OBJ);
    _Float16* row = (_Float16*)(smem + X_OFF) + tid * 32;
#pragma unroll
    for (int i = 0; i < 10; ++i) row[i] = (_Float16)ob[i];
#pragma unroll
    for (int i = 0; i < 3; ++i) {
      int idx = pred_ids[p * 3 + i];
      row[10 + i] = (_Float16)(g[(size_t)b * 30 + idx] - ag[(size_t)b * 30 + idx]);
    }
    int e0 = edges[p * 2], e1 = edges[p * 2 + 1];
#pragma unroll
    for (int i = 0; i < 3; ++i) row[13 + i] = (_Float16)ob[10 + 15 * e0 + i];
#pragma unroll
    for (int i = 0; i < 3; ++i) row[16 + i] = (_Float16)ob[10 + 15 * e1 + i];
#pragma unroll
    for (int i = 19; i < 32; ++i) row[i] = (_Float16)0.0f;
  }
  __syncthreads();

  // ---- stage 1: h = relu(inp @ W1 + b1)   (M=160, K=32, N=256) ----
  for (int t = w; t < 10 * 16; t += 8) {
    int mt = t >> 4, nt = t & 15;
    const _Float16* arow = (const _Float16*)(smem + X_OFF) + (mt * 16 + (lane & 15)) * 32;
    v16h a = load_a(arow, 0, lane);
    v16h bf = *(const v16h*)(wf + WF_W1 + (size_t)nt * 512 + lane * 16);
    v8f acc = {};
    acc = wmma16(a, bf, acc);
    float bias = b1[nt * 16 + (lane & 15)];
    _Float16* hc = (_Float16*)(smem + H_OFF) + nt * 16 + (lane & 15);
    int mb = mt * 16 + ((lane >> 4) & 1) * 8;
#pragma unroll
    for (int r = 0; r < 8; ++r) {
      float v = acc[r] + bias; v = v > 0.f ? v : 0.f;
      hc[(mb + r) * 256] = (_Float16)v;
    }
  }
  __syncthreads();

  // ---- stage 2: x = relu(h @ W2 + b2)    (M=160, K=256, N=128) ----
  for (int t = w; t < 10 * 8; t += 8) {
    int mt = t >> 3, nt = t & 7;
    v8f acc = {};
    const _Float16* arow = (const _Float16*)(smem + H_OFF) + (mt * 16 + (lane & 15)) * 256;
#pragma unroll
    for (int kt = 0; kt < 8; ++kt) {
      v16h a = load_a(arow, kt * 32, lane);
      v16h bf = *(const v16h*)(wf + WF_W2 + (size_t)(kt * 8 + nt) * 512 + lane * 16);
      acc = wmma16(a, bf, acc);
    }
    float bias = b2[nt * 16 + (lane & 15)];
    _Float16* xc = (_Float16*)(smem + X_OFF) + nt * 16 + (lane & 15);
    int mb = mt * 16 + ((lane >> 4) & 1) * 8;
#pragma unroll
    for (int r = 0; r < 8; ++r) {
      float v = acc[r] + bias; v = v > 0.f ? v : 0.f;
      xc[(mb + r) * 128] = (_Float16)v;
    }
  }
  __syncthreads();

  // ---- stage 3: zero pads (Q/K rows 20..31, V^T cols 20..31), then QKV ----
  for (int i = tid; i < 8 * 12 * 2 * 64; i += 256) {       // Q,K pad rows (dwords)
    int buf = i / (8 * 12 * 64); int rem = i % (8 * 12 * 64);
    int bb = rem / (12 * 64); int rr = (rem / 64) % 12; int dw = rem % 64;
    ((unsigned int*)(smem + (buf ? K_OFF : H_OFF)))[(bb * 32 + 20 + rr) * 64 + dw] = 0u;
  }
  for (int i = tid; i < 8 * 128 * 6; i += 256) {           // V^T pad (edges 20..31)
    int seg = i / 6, dw = i % 6;
    ((unsigned int*)(smem + VT_OFF))[seg * 16 + 10 + dw] = 0u;
  }
  for (int t = w; t < 3 * 80; t += 8) {                    // Q,K,V GEMMs
    int mat = t / 80, tt = t % 80;
    int mt = tt >> 3, nt = tt & 7;
    const _Float16* Wf = wf + WF_WQ + (size_t)mat * 16384;
    const float* bp = (mat == 0) ? bq : (mat == 1) ? bk : bv;
    v8f acc = {};
    const _Float16* arow = (const _Float16*)(smem + X_OFF) + (mt * 16 + (lane & 15)) * 128;
#pragma unroll
    for (int kt = 0; kt < 4; ++kt) {
      v16h a = load_a(arow, kt * 32, lane);
      v16h bf = *(const v16h*)(Wf + (size_t)(kt * 8 + nt) * 512 + lane * 16);
      acc = wmma16(a, bf, acc);
    }
    int n = nt * 16 + (lane & 15);
    float bias = bp[n];
    int mb = mt * 16 + ((lane >> 4) & 1) * 8;
#pragma unroll
    for (int r = 0; r < 8; ++r) {
      int m = mb + r;                       // global edge-row 0..159
      int bb = m / NPERM, e = m % NPERM;
      float v = acc[r] + bias;
      if (mat == 0) {                       // Q (pre-scaled by 1/sqrt(128))
        v *= 0.08838834764831845f;
        ((_Float16*)(smem + H_OFF))[(bb * 32 + e) * 128 + n] = (_Float16)v;
      } else if (mat == 1) {                // K
        ((_Float16*)(smem + K_OFF))[(bb * 32 + e) * 128 + n] = (_Float16)v;
      } else {                              // V transposed: [b][n][edge]
        ((_Float16*)(smem + VT_OFF))[(bb * 128 + n) * 32 + e] = (_Float16)v;
      }
    }
  }
  __syncthreads();

  // ---- stage 4: scores = Q K^T (per batch 32x32 padded, K=128); also zero attn/pool pads ----
  for (int i = tid; i < 4096; i += 256) ((unsigned int*)(smem + AT_OFF))[i] = 0u;   // attn tiles
  for (int i = tid; i < 512; i += 256) ((unsigned int*)(smem + PA_OFF))[512 + i] = 0u; // pooled rows 8..15
  for (int t = w; t < 32; t += 8) {
    int bb = t >> 2, mt = (t >> 1) & 1, nt = t & 1;
    v8f acc = {};
    const _Float16* qrow = (const _Float16*)(smem + H_OFF) + (bb * 32 + mt * 16 + (lane & 15)) * 128;
    const _Float16* krow = (const _Float16*)(smem + K_OFF) + (bb * 32 + nt * 16 + (lane & 15)) * 128;
#pragma unroll
    for (int kt = 0; kt < 4; ++kt) {
      v16h a = load_a(qrow, kt * 32, lane);
      v16h bf = load_b_rows(krow, kt * 32, lane);
      acc = wmma16(a, bf, acc);
    }
    int n = nt * 16 + (lane & 15);
    int mb = mt * 16 + ((lane >> 4) & 1) * 8;
    if (n < NPERM) {
      float* sc = (float*)(smem + SC_OFF) + bb * 400;
#pragma unroll
      for (int r = 0; r < 8; ++r) { int m = mb + r; if (m < NPERM) sc[m * 20 + n] = acc[r]; }
    }
  }
  __syncthreads();

  // ---- stage 5: softmax over 20, write zero-padded f16 attn A-tiles ----
  if (tid < MROWS) {
    int bb = tid / NPERM, m = tid % NPERM;
    const float* sc = (const float*)(smem + SC_OFF) + bb * 400 + m * 20;
    float mx = sc[0];
#pragma unroll
    for (int i = 1; i < 20; ++i) mx = fmaxf(mx, sc[i]);
    float e[20], s = 0.f;
#pragma unroll
    for (int i = 0; i < 20; ++i) { e[i] = __expf(sc[i] - mx); s += e[i]; }
    float inv = 1.f / s;
    _Float16* ar = (_Float16*)(smem + AT_OFF) + (bb * 32 + m) * 32;
#pragma unroll
    for (int i = 0; i < 20; ++i) ar[i] = (_Float16)(e[i] * inv);
  }
  __syncthreads();

  // ---- stage 6: attn @ V, pooled over edges (column sums) ----
  for (int t = w; t < 64; t += 8) {
    int bb = t >> 3, nt = t & 7;
    const _Float16* vrow = (const _Float16*)(smem + VT_OFF) + (bb * 128 + nt * 16 + (lane & 15)) * 32;
    v16h bf = load_b_rows(vrow, 0, lane);
    v8f acc = {};
#pragma unroll
    for (int mt = 0; mt < 2; ++mt) {   // fold both M-tiles into one acc: we only need column sums
      const _Float16* arow = (const _Float16*)(smem + AT_OFF) + (bb * 32 + mt * 16 + (lane & 15)) * 32;
      v16h a = load_a(arow, 0, lane);
      acc = wmma16(a, bf, acc);
    }
    float s = 0.f;
#pragma unroll
    for (int r = 0; r < 8; ++r) s += acc[r];
    s += __shfl_xor(s, 16, 32);
    if (lane < 16) {
      ((_Float16*)(smem + PA_OFF))[bb * 128 + nt * 16 + lane] = (_Float16)s;
    }
  }
  __syncthreads();

  // ---- stage 7: hr = relu(pooled @ Wr + br)  (M=16 pad(8 real), K=128, N=256) ----
  for (int t = w; t < 16; t += 8) {
    int nt = t;
    v8f acc = {};
    const _Float16* arow = (const _Float16*)(smem + PA_OFF) + (lane & 15) * 128;
#pragma unroll
    for (int kt = 0; kt < 4; ++kt) {
      v16h a = load_a(arow, kt * 32, lane);
      v16h bf = *(const v16h*)(wf + WF_WR + (size_t)(kt * 16 + nt) * 512 + lane * 16);
      acc = wmma16(a, bf, acc);
    }
    float bias = br[nt * 16 + (lane & 15)];
    _Float16* hc = (_Float16*)(smem + HR_OFF) + nt * 16 + (lane & 15);
    int mb = ((lane >> 4) & 1) * 8;
#pragma unroll
    for (int r = 0; r < 8; ++r) {
      float v = acc[r] + bias; v = v > 0.f ? v : 0.f;
      hc[(mb + r) * 256] = (_Float16)v;
    }
  }
  __syncthreads();

  // ---- stage 8: heads (N=4 each, scalar dot products) ----
  if (tid < 64) {
    int bi = tid >> 3, j = tid & 7;
    int col = j & 3; bool is_std = (j >= 4);
    const float* Wp = is_std ? Ws : Wm;
    const float* bp = is_std ? bs : bm;
    const _Float16* hr = (const _Float16*)(smem + HR_OFF) + bi * 256;
    float s = bp[col];
#pragma unroll 8
    for (int i = 0; i < 256; ++i) s += (float)hr[i] * Wp[i * 4 + col];
    if (is_std) s = fminf(fmaxf(s, -20.f), 2.f);
    int b = blockIdx.x * TB + bi;
    out[(is_std ? (size_t)BATCH * 4 : 0) + (size_t)b * 4 + col] = s;
  }
}

extern "C" void kernel_launch(void* const* d_in, const int* in_sizes, int n_in,
                              void* d_out, int out_size, void* d_ws, size_t ws_size,
                              hipStream_t stream) {
  const float* obs = (const float*)d_in[0];
  const float* ag  = (const float*)d_in[1];
  const float* g   = (const float*)d_in[2];
  const float* W1  = (const float*)d_in[3];  const float* b1 = (const float*)d_in[4];
  const float* W2  = (const float*)d_in[5];  const float* b2 = (const float*)d_in[6];
  const float* Wq  = (const float*)d_in[7];  const float* bq = (const float*)d_in[8];
  const float* Wk  = (const float*)d_in[9];  const float* bk = (const float*)d_in[10];
  const float* Wv  = (const float*)d_in[11]; const float* bv = (const float*)d_in[12];
  const float* Wr  = (const float*)d_in[13]; const float* br = (const float*)d_in[14];
  const float* Wm  = (const float*)d_in[15]; const float* bm = (const float*)d_in[16];
  const float* Ws  = (const float*)d_in[17]; const float* bs = (const float*)d_in[18];
  const int* edges    = (const int*)d_in[19];
  const int* pred_ids = (const int*)d_in[20];
  _Float16* wf = (_Float16*)d_ws;   // needs 245760 bytes

  pack_weights<<<16, 32, 0, stream>>>(W1, wf + WF_W1, 19, 256, 16);
  pack_weights<<<64, 32, 0, stream>>>(W2, wf + WF_W2, 256, 128, 8);
  pack_weights<<<32, 32, 0, stream>>>(Wq, wf + WF_WQ, 128, 128, 8);
  pack_weights<<<32, 32, 0, stream>>>(Wk, wf + WF_WK, 128, 128, 8);
  pack_weights<<<32, 32, 0, stream>>>(Wv, wf + WF_WV, 128, 128, 8);
  pack_weights<<<64, 32, 0, stream>>>(Wr, wf + WF_WR, 128, 256, 16);

  actor_fused<<<BATCH / TB, 256, 0, stream>>>(
      obs, ag, g, b1, b2, bq, bk, bv, br, Wm, bm, Ws, bs,
      edges, pred_ids, wf, (float*)d_out);
}
